// GATE_89876485636587
// MI455X (gfx1250) — matrix operand order repaired and checked by reading, not compiled, not run on hardware
//
#include <hip/hip_runtime.h>
#include <hip/hip_bf16.h>

// ---------------------------------------------------------------------------
// 2-layer GATConv (heads=1, self-loops) for MI455X (gfx1250, wave32).
// Dense transforms use v_wmma_f32_16x16x32_bf16 with weights staged in LDS in
// fragment-major (column-panel) order so each lane loads its whole B fragment
// as one contiguous v16bf (2x ds_load_b128). Sparse softmax/scatter uses
// coalesced lane-per-feature float atomics (memory-bound).
// ---------------------------------------------------------------------------

typedef __attribute__((ext_vector_type(16))) __bf16 v16bf;
typedef __attribute__((ext_vector_type(8)))  float  v8f;

#define NEG_SLOPE 0.2f

// ---------------- helpers ----------------

__device__ __forceinline__ float lrelu(float x) { return x >= 0.f ? x : NEG_SLOPE * x; }

// Order-preserving float <-> int encoding so we can use integer atomicMax.
__device__ __forceinline__ int f2ord(float f) {
    int i = __float_as_int(f);
    return i >= 0 ? i : (i ^ 0x7FFFFFFF);
}
__device__ __forceinline__ float ord2f(int i) {
    return __int_as_float(i >= 0 ? i : (i ^ 0x7FFFFFFF));
}

__device__ __forceinline__ void edge_ends(const int* __restrict__ ei, int E, int e,
                                          int& s, int& d) {
    if (e < E) { s = ei[e]; d = ei[(size_t)E + e]; }
    else       { s = d = e - E; }            // appended self-loop
}

// ---------------- init ----------------

__global__ void k_init(int* __restrict__ m, float* __restrict__ den,
                       float* __restrict__ agg, int n, long aggCount) {
    long i = (long)blockIdx.x * blockDim.x + threadIdx.x;
    if (i < n) { m[i] = (int)0x80000000; den[i] = 0.f; }
    if (i < aggCount) agg[i] = 0.f;
}

// ---------------- GEMM layer 1: h1 = x[N,128] @ W1[128,32]  (bf16 WMMA) ----
//
// LDS holds W1 as 256 B-fragments of 16 bf16 each (8 KB total):
//   fragment id f = (s*2 + khalf)*32 + col,  element e = k - (32s + 16*khalf)
//   lw[f][e] = bf16( W1[(32s + 16*khalf + e)*32 + col] )
// so a lane's whole B fragment is a single contiguous 32-byte v16bf.

__global__ void k_gemm1(const float* __restrict__ x, const float* __restrict__ W,
                        float* __restrict__ h, int n) {
    __shared__ v16bf lw[256];                // fragment-major W1 (bf16, 8 KB)
    {
        __bf16* lwh = (__bf16*)lw;
        for (int i = threadIdx.x; i < 128 * 32; i += blockDim.x) {
            const int f   = i >> 4;          // fragment id
            const int e   = i & 15;          // element within fragment
            const int col = f & 31;          // output column 0..31
            const int sh  = f >> 5;          // s*2 + khalf
            const int k   = (sh << 4) + e;   // 32s + 16*khalf + e
            lwh[i] = (__bf16)W[(size_t)k * 32 + col];
        }
    }
    __syncthreads();

    const int wave    = blockIdx.x * (blockDim.x >> 5) + (threadIdx.x >> 5);
    const int lane    = threadIdx.x & 31;
    const int rowBase = wave * 16;
    if (rowBase >= n) return;                // whole-wave uniform exit (EXEC stays all-1s)

    const int mrow  = lane & 15;
    const int khalf = lane >> 4;             // 0: lanes 0-15, 1: lanes 16-31
    int arow = rowBase + mrow;
    if (arow >= n) arow = n - 1;             // clamp keeps every lane active

    v8f acc0 = {}, acc1 = {};
    for (int s = 0; s < 4; ++s) {            // K = 128 in 4 steps of 32
        const int kb = s * 32;
        if (s < 3) __builtin_prefetch(x + (size_t)arow * 128 + kb + 32, 0, 3);

        // A fragment: 16x32 bf16, row = lane&15; per lane one contiguous
        // 16-float run (kb+16*khalf) -> global_load_b128 x4
        v16bf a;
#pragma unroll
        for (int v = 0; v < 8; ++v) {
            const int kk = kb + 2 * v + khalf * 16;
            const float2 t = *(const float2*)(x + (size_t)arow * 128 + kk);
            a[2 * v]     = (__bf16)t.x;
            a[2 * v + 1] = (__bf16)t.y;
        }
        // B fragments: one contiguous v16bf per lane from LDS
        const int fb = (s * 2 + khalf) * 32 + (lane & 15);
        const v16bf b0 = lw[fb];
        const v16bf b1 = lw[fb + 16];

        acc0 = __builtin_amdgcn_wmma_f32_16x16x32_bf16(false, a, false, b0,
                                                       (short)0, acc0, false, false);
        acc1 = __builtin_amdgcn_wmma_f32_16x16x32_bf16(false, a, false, b1,
                                                       (short)0, acc1, false, false);
    }
    // C/D layout: VGPR r -> M = r + 8*khalf, N = lane&15 (acc0) / 16+lane&15 (acc1)
#pragma unroll
    for (int r = 0; r < 8; ++r) {
        const int row = rowBase + r + 8 * khalf;
        if (row < n) {
            h[(size_t)row * 32 + (lane & 15)]      = acc0[r];
            h[(size_t)row * 32 + 16 + (lane & 15)] = acc1[r];
        }
    }
}

// ---------------- GEMM layer 2: h2 = h1r[N,32] @ W2[32,16] (one WMMA/tile) --
// Same fragment-major staging: 32 fragments (khalf,col) of 16 bf16 (1 KB).

__global__ void k_gemm2(const float* __restrict__ hin, const float* __restrict__ W,
                        float* __restrict__ hout, int n) {
    __shared__ v16bf lw[32];
    {
        __bf16* lwh = (__bf16*)lw;
        for (int i = threadIdx.x; i < 32 * 16; i += blockDim.x) {
            const int f   = i >> 4;          // khalf*16 + col
            const int e   = i & 15;
            const int col = f & 15;
            const int kh  = f >> 4;
            lwh[i] = (__bf16)W[(size_t)(kh * 16 + e) * 16 + col];
        }
    }
    __syncthreads();

    const int wave    = blockIdx.x * (blockDim.x >> 5) + (threadIdx.x >> 5);
    const int lane    = threadIdx.x & 31;
    const int rowBase = wave * 16;
    if (rowBase >= n) return;

    const int mrow  = lane & 15;
    const int khalf = lane >> 4;
    int arow = rowBase + mrow;
    if (arow >= n) arow = n - 1;

    v16bf a;
#pragma unroll
    for (int v = 0; v < 8; ++v) {
        const int kk = 2 * v + khalf * 16;
        const float2 t = *(const float2*)(hin + (size_t)arow * 32 + kk);
        a[2 * v]     = (__bf16)t.x;
        a[2 * v + 1] = (__bf16)t.y;
    }
    const v16bf b = lw[khalf * 16 + (lane & 15)];

    v8f acc = {};
    acc = __builtin_amdgcn_wmma_f32_16x16x32_bf16(false, a, false, b,
                                                  (short)0, acc, false, false);
#pragma unroll
    for (int r = 0; r < 8; ++r) {
        const int row = rowBase + r + 8 * khalf;
        if (row < n) hout[(size_t)row * 16 + (lane & 15)] = acc[r];
    }
}

// ---------------- per-node attention dots: al_s = h@a_src, al_d = h@a_dst ---

template <int C>
__global__ void k_dots(const float* __restrict__ h, const float* __restrict__ asrc,
                       const float* __restrict__ adst, float* __restrict__ als,
                       float* __restrict__ ald, int n) {
    const int i = blockIdx.x * blockDim.x + threadIdx.x;
    if (i >= n) return;
    float s = 0.f, d = 0.f;
#pragma unroll
    for (int c = 0; c < C; ++c) {
        const float v = h[(size_t)i * C + c];
        s += v * asrc[c];
        d += v * adst[c];
    }
    als[i] = s;
    ald[i] = d;
}

// ---------------- edge pass 1: segment max of leaky-relu attention ----------

__global__ void k_amax(const int* __restrict__ ei, int E, int ET,
                       const float* __restrict__ als, const float* __restrict__ ald,
                       int* __restrict__ m) {
    const int e = blockIdx.x * blockDim.x + threadIdx.x;
    if (e >= ET) return;
    int s, d;
    edge_ends(ei, E, e, s, d);
    atomicMax(&m[d], f2ord(lrelu(als[s] + ald[d])));
}

// ---------------- edge pass 2: e = exp(alpha - max), segment sum ------------

__global__ void k_expsum(const int* __restrict__ ei, int E, int ET,
                         const float* __restrict__ als, const float* __restrict__ ald,
                         const int* __restrict__ m, float* __restrict__ ebuf,
                         float* __restrict__ den) {
    const int e = blockIdx.x * blockDim.x + threadIdx.x;
    if (e >= ET) return;
    int s, d;
    edge_ends(ei, E, e, s, d);
    const float a  = lrelu(als[s] + ald[d]);
    const float ex = __expf(a - ord2f(m[d]));
    ebuf[e] = ex;
    atomicAdd(&den[d], ex);
}

// ---------------- edge pass 3: agg[dst] += coef * h[src] (lane per feature) -

template <int C>
__global__ void k_scatter(const int* __restrict__ ei, int E, int ET,
                          const float* __restrict__ ebuf, const float* __restrict__ den,
                          const float* __restrict__ h, float* __restrict__ agg) {
    const long tid = (long)blockIdx.x * blockDim.x + threadIdx.x;
    const int  e   = (int)(tid / C);         // C contiguous lanes share one edge
    const int  c   = (int)(tid % C);         // -> coalesced gather + atomics
    if (e >= ET) return;
    int s, d;
    edge_ends(ei, E, e, s, d);
    const float coef = ebuf[e] / (den[d] + 1e-16f);
    atomicAdd(&agg[(size_t)d * C + c], coef * h[(size_t)s * C + c]);
}

// ---------------- epilogues ----------------

__global__ void k_bias_relu32(const float* __restrict__ agg, const float* __restrict__ b,
                              float* __restrict__ out, long total) {
    const long i = (long)blockIdx.x * blockDim.x + threadIdx.x;
    if (i >= total) return;
    const float v = agg[i] + b[i & 31];
    out[i] = v > 0.f ? v : 0.f;
}

__global__ void k_bias16_inplace(float* __restrict__ out, const float* __restrict__ b,
                                 long total) {
    const long i = (long)blockIdx.x * blockDim.x + threadIdx.x;
    if (i >= total) return;
    out[i] += b[i & 15];
}

// ---------------------------------------------------------------------------

extern "C" void kernel_launch(void* const* d_in, const int* in_sizes, int n_in,
                              void* d_out, int out_size, void* d_ws, size_t ws_size,
                              hipStream_t stream) {
    (void)n_in; (void)out_size; (void)ws_size;

    const float* x   = (const float*)d_in[0];
    const int*   ei  = (const int*)  d_in[1];
    /* d_in[2] = ew : unused by GATConv (edge_dim=None) */
    const float* W1  = (const float*)d_in[3];
    const float* as1 = (const float*)d_in[4];
    const float* ad1 = (const float*)d_in[5];
    const float* b1  = (const float*)d_in[6];
    const float* W2  = (const float*)d_in[7];
    const float* as2 = (const float*)d_in[8];
    const float* ad2 = (const float*)d_in[9];
    const float* b2  = (const float*)d_in[10];
    float* out = (float*)d_out;

    const int n  = in_sizes[0] / 128;
    const int E  = in_sizes[1] / 2;
    const int ET = E + n;

    // workspace layout (floats); heavy buffers are reused across layers
    float* ws   = (float*)d_ws;
    size_t off  = 0;
    float* h1   = ws + off; off += (size_t)n * 32;  // layer1 features, later relu'd copy
    float* agg1 = ws + off; off += (size_t)n * 32;  // layer1 aggregate, later h2
    float* als  = ws + off; off += n;
    float* ald  = ws + off; off += n;
    int*   m    = (int*)(ws + off); off += n;
    float* den  = ws + off; off += n;
    float* ebuf = ws + off; off += ET;

    const int B = 256;
    const int gNode  = (n + B - 1) / B;
    const int gEdge  = (ET + B - 1) / B;
    const int gTiles = (((n + 15) / 16) + 7) / 8;   // 8 waves (wave32) per block

    // ----- layer 1 -----
    {
        const long cnt = (long)n * 32;
        k_init<<<(int)((cnt + B - 1) / B), B, 0, stream>>>(m, den, agg1, n, cnt);
    }
    k_gemm1<<<gTiles, B, 0, stream>>>(x, W1, h1, n);
    k_dots<32><<<gNode, B, 0, stream>>>(h1, as1, ad1, als, ald, n);
    k_amax<<<gEdge, B, 0, stream>>>(ei, E, ET, als, ald, m);
    k_expsum<<<gEdge, B, 0, stream>>>(ei, E, ET, als, ald, m, ebuf, den);
    {
        const long t = (long)ET * 32;
        k_scatter<32><<<(int)((t + B - 1) / B), B, 0, stream>>>(ei, E, ET, ebuf, den, h1, agg1);
    }
    {
        const long t = (long)n * 32;   // h1 := relu(agg1 + b1)
        k_bias_relu32<<<(int)((t + B - 1) / B), B, 0, stream>>>(agg1, b1, h1, t);
    }

    // ----- layer 2 (accumulate straight into d_out) -----
    {
        const long cnt = (long)n * 16;
        k_init<<<(int)((cnt + B - 1) / B), B, 0, stream>>>(m, den, out, n, cnt);
    }
    k_gemm2<<<gTiles, B, 0, stream>>>(h1, W2, agg1, n);       // h2 lives in agg1
    k_dots<16><<<gNode, B, 0, stream>>>(agg1, as2, ad2, als, ald, n);
    k_amax<<<gEdge, B, 0, stream>>>(ei, E, ET, als, ald, m);
    k_expsum<<<gEdge, B, 0, stream>>>(ei, E, ET, als, ald, m, ebuf, den);
    {
        const long t = (long)ET * 16;
        k_scatter<16><<<(int)((t + B - 1) / B), B, 0, stream>>>(ei, E, ET, ebuf, den, agg1, out);
    }
    {
        const long t = (long)n * 16;
        k_bias16_inplace<<<(int)((t + B - 1) / B), B, 0, stream>>>(out, b2, t);
    }
}
